// LoRALinear_3530463117956
// MI455X (gfx1250) — matrix operand order repaired and checked by reading, not compiled, not run on hardware
//
#include <hip/hip_runtime.h>

// Problem constants (match reference)
#define B_SZ      16384
#define IN_F      1024
#define OUT_F     1024
#define RANK      8
#define N_DOM     64

// GEMM tiling
#define BM        128      // block tile rows (batch)
#define BN        128      // block tile cols (out features)
#define BK        32       // K slice (one bf16 WMMA K)
#define LDA       40       // padded LDS row stride (halves) -> conflict-free b128 reads

typedef __bf16 v16bf __attribute__((ext_vector_type(16)));
typedef float  v8f   __attribute__((ext_vector_type(8)));

struct __attribute__((aligned(16))) U128 { unsigned int a, b, c, d; };
struct U256 { U128 lo, hi; };

__device__ __forceinline__ v16bf frag_from(const unsigned short* p0,
                                           const unsigned short* p1) {
  U256 u;
  u.lo = *reinterpret_cast<const U128*>(p0);   // ds_load_b128
  u.hi = *reinterpret_cast<const U128*>(p1);   // ds_load_b128
  return __builtin_bit_cast(v16bf, u);
}

__device__ __forceinline__ unsigned short bf16_rn(float f) {
  unsigned u = __float_as_uint(f);
  unsigned r = u + 0x7FFFu + ((u >> 16) & 1u);   // round-to-nearest-even
  return (unsigned short)(r >> 16);
}

// split two fp32 into packed bf16 hi / bf16 lo (residual) words
__device__ __forceinline__ void split2(float f0, float f1,
                                       unsigned& hp, unsigned& lp) {
  unsigned short h0 = bf16_rn(f0), h1 = bf16_rn(f1);
  float r0 = f0 - __uint_as_float((unsigned)h0 << 16);
  float r1 = f1 - __uint_as_float((unsigned)h1 << 16);
  unsigned short l0 = bf16_rn(r0), l1 = bf16_rn(r1);
  hp = (unsigned)h0 | ((unsigned)h1 << 16);
  lp = (unsigned)l0 | ((unsigned)l1 << 16);
}

// ---------------------------------------------------------------------------
// Kernel 1: L[d,o] = sum_r (sum_i A[d,r,i]) * B[d,r,o]      (the LoRA term
// does not depend on x at all -- only on domain_id)
// ---------------------------------------------------------------------------
__global__ void lora_domain_table_kernel(const float* __restrict__ A_table,
                                         const float* __restrict__ B_table,
                                         float* __restrict__ L) {
  const int d    = blockIdx.x;        // 64 domains
  const int t    = threadIdx.x;       // 256 threads = 8 waves
  const int wave = t >> 5;            // wave handles rank r = wave
  const int lane = t & 31;

  __shared__ float s_r[RANK];

  const float* Arow = A_table + (size_t)d * RANK * IN_F + (size_t)wave * IN_F;
  float acc = 0.f;
  for (int i = lane; i < IN_F; i += 32) acc += Arow[i];
#pragma unroll
  for (int off = 16; off > 0; off >>= 1) acc += __shfl_down(acc, off, 32);
  if (lane == 0) s_r[wave] = acc;
  __syncthreads();

  float s[RANK];
#pragma unroll
  for (int r = 0; r < RANK; ++r) s[r] = s_r[r];

  const float* Bd = B_table + (size_t)d * RANK * OUT_F;
  for (int o = t; o < OUT_F; o += 256) {
    float v = 0.f;
#pragma unroll
    for (int r = 0; r < RANK; ++r) v += s[r] * Bd[r * OUT_F + o];
    L[d * OUT_F + o] = v;
  }
}

// ---------------------------------------------------------------------------
// Kernel 2: split W (fp32, row-major [out_f][in_f]) into bf16 hi/lo once.
// Done once (4 MB) instead of per-row-block in the GEMM; lets the GEMM
// stage W tiles with async global->LDS copies (no conversion VALU).
// ---------------------------------------------------------------------------
__global__ void wsplit_kernel(const float* __restrict__ Wm,
                              unsigned short* __restrict__ Wh,
                              unsigned short* __restrict__ Wl) {
  const int idx = blockIdx.x * 256 + threadIdx.x;   // 1 float4 per thread
  const float4 v = *reinterpret_cast<const float4*>(&Wm[(size_t)idx * 4]);
  unsigned h01, l01, h23, l23;
  split2(v.x, v.y, h01, l01);
  split2(v.z, v.w, h23, l23);
  *reinterpret_cast<unsigned*>(&Wh[(size_t)idx * 4])     = h01;
  *reinterpret_cast<unsigned*>(&Wh[(size_t)idx * 4 + 2]) = h23;
  *reinterpret_cast<unsigned*>(&Wl[(size_t)idx * 4])     = l01;
  *reinterpret_cast<unsigned*>(&Wl[(size_t)idx * 4 + 2]) = l23;
}

// ---------------------------------------------------------------------------
// Kernel 3: out = x @ W^T + L[dom]   via bf16x3 split-precision WMMA.
// Block: 256 threads (8 waves), 128x128 output tile, K staged 32 at a time.
// W tiles (pre-split bf16) are staged with GLOBAL_LOAD_ASYNC_TO_LDS_B128
// (ASYNCcnt), overlapping the x-tile fp32->bf16 split done on the VALU.
// ---------------------------------------------------------------------------
__global__ void __launch_bounds__(256)
lora_linear_gemm_kernel(const float* __restrict__ x,
                        const unsigned short* __restrict__ Wh,
                        const unsigned short* __restrict__ Wl,
                        const int*   __restrict__ dom,
                        const float* __restrict__ Lt,
                        float*       __restrict__ out) {
  __shared__ unsigned short As_h[BM * LDA];
  __shared__ unsigned short As_l[BM * LDA];
  __shared__ unsigned short Bs_h[BN * LDA];
  __shared__ unsigned short Bs_l[BN * LDA];

  const int t       = threadIdx.x;
  const int lane    = t & 31;
  const int wave    = t >> 5;
  const int wm      = wave >> 1;        // 0..3  (row group of 32)
  const int wn      = wave & 1;         // 0..1  (col group of 64)
  const int g       = lane >> 4;        // half-wave group
  const int ml      = lane & 15;        // row (A) / col (B) within tile
  const int rowBase = blockIdx.y * BM;  // batch rows
  const int colBase = blockIdx.x * BN;  // out features

  v8f acc[2][4] = {};                   // fp32 accumulators, 8 tiles/wave

  for (int k0 = 0; k0 < IN_F; k0 += BK) {
    __syncthreads();                    // previous compute done before refill

    // ---- W tiles: async bf16 copies global -> LDS (tracked by ASYNCcnt).
    // Generic LDS addresses carry the wave-relative LDS offset in addr[31:0]
    // (ISA aperture rules), so the low 32 bits form the VDST LDS address.
#pragma unroll
    for (int j = 0; j < 2; ++j) {
      const int idx = j * 256 + t;      // 512 x 16-byte chunks per tile
      const int r   = idx >> 2;         // 0..127  (W row = out feature)
      const int c8  = (idx & 3) << 3;   // 0,8,16,24 (bf16 col)
      const size_t goff = (size_t)(colBase + r) * IN_F + k0 + c8;
      const unsigned long long gah = (unsigned long long)(uintptr_t)(&Wh[goff]);
      const unsigned long long gal = (unsigned long long)(uintptr_t)(&Wl[goff]);
      const unsigned ldsh = (unsigned)(uintptr_t)(&Bs_h[r * LDA + c8]);
      const unsigned ldsl = (unsigned)(uintptr_t)(&Bs_l[r * LDA + c8]);
      asm volatile("global_load_async_to_lds_b128 %0, %1, off"
                   :: "v"(ldsh), "v"(gah) : "memory");
      asm volatile("global_load_async_to_lds_b128 %0, %1, off"
                   :: "v"(ldsl), "v"(gal) : "memory");
    }

    // ---- x tile: global fp32 -> bf16 hi/lo split in LDS (overlaps async)
#pragma unroll
    for (int j = 0; j < 4; ++j) {
      const int idx = j * 256 + t;      // 1024 float4 slots per tile
      const int r   = idx >> 3;         // tile row (0..127)
      const int c   = (idx & 7) << 2;   // tile col (0,4,..,28)
      const float4 vx = *reinterpret_cast<const float4*>(
          &x[(size_t)(rowBase + r) * IN_F + k0 + c]);
      unsigned h01, l01, h23, l23;
      split2(vx.x, vx.y, h01, l01);
      split2(vx.z, vx.w, h23, l23);
      *reinterpret_cast<unsigned*>(&As_h[r * LDA + c])     = h01;
      *reinterpret_cast<unsigned*>(&As_h[r * LDA + c + 2]) = h23;
      *reinterpret_cast<unsigned*>(&As_l[r * LDA + c])     = l01;
      *reinterpret_cast<unsigned*>(&As_l[r * LDA + c + 2]) = l23;
    }
    // prefetch next x K-slice (global_prefetch_b8 path)
    if (k0 + BK < IN_F) {
      __builtin_prefetch(&x[(size_t)(rowBase + (t >> 1)) * IN_F + k0 + BK], 0, 1);
    }

    asm volatile("s_wait_asynccnt 0" ::: "memory");  // our LDS writes landed
    __syncthreads();                                  // everyone's landed

    // ---- A fragments (16x32 bf16): lane = row, VGPR0-3 K=g*8.., VGPR4-7 K=16+g*8..
    v16bf Ah[2], Al[2];
#pragma unroll
    for (int mt = 0; mt < 2; ++mt) {
      const int rl   = wm * 32 + mt * 16 + ml;
      const int base = rl * LDA + g * 8;
      Ah[mt] = frag_from(&As_h[base], &As_h[base + 16]);
      Al[mt] = frag_from(&As_l[base], &As_l[base + 16]);
    }

    // ---- B fragments (32x16 bf16): lane = col, VGPR0-3 K=g*8.., VGPR4-7 K=16+g*8..
#pragma unroll
    for (int nt = 0; nt < 4; ++nt) {
      const int cl   = wn * 64 + nt * 16 + ml;
      const int base = cl * LDA + g * 8;
      const v16bf Bh = frag_from(&Bs_h[base], &Bs_h[base + 16]);
      const v16bf Bl = frag_from(&Bs_l[base], &Bs_l[base + 16]);
#pragma unroll
      for (int mt = 0; mt < 2; ++mt) {
        acc[mt][nt] = __builtin_amdgcn_wmma_f32_16x16x32_bf16(
            false, Ah[mt], false, Bh, (short)0, acc[mt][nt], false, false);
        acc[mt][nt] = __builtin_amdgcn_wmma_f32_16x16x32_bf16(
            false, Ah[mt], false, Bl, (short)0, acc[mt][nt], false, false);
        acc[mt][nt] = __builtin_amdgcn_wmma_f32_16x16x32_bf16(
            false, Al[mt], false, Bh, (short)0, acc[mt][nt], false, false);
      }
    }
  }

  // ---- epilogue: out[row,col] = acc + L[dom[row], col]
  // C/D layout: VGPR j, lanes 0-15 -> M=j, lanes 16-31 -> M=8+j; N = lane&15
#pragma unroll
  for (int mt = 0; mt < 2; ++mt) {
    const int rowb = rowBase + wm * 32 + mt * 16 + g * 8;
    int doms[8];
#pragma unroll
    for (int j = 0; j < 8; ++j) doms[j] = dom[rowb + j];
#pragma unroll
    for (int nt = 0; nt < 4; ++nt) {
      const int col = colBase + wn * 64 + nt * 16 + ml;
#pragma unroll
      for (int j = 0; j < 8; ++j) {
        out[(size_t)(rowb + j) * OUT_F + col] =
            acc[mt][nt][j] + Lt[doms[j] * OUT_F + col];
      }
    }
  }
}

// ---------------------------------------------------------------------------
extern "C" void kernel_launch(void* const* d_in, const int* in_sizes, int n_in,
                              void* d_out, int out_size, void* d_ws, size_t ws_size,
                              hipStream_t stream) {
  const float* x       = (const float*)d_in[0];   // [16384,1024]
  const float* Wm      = (const float*)d_in[1];   // [1024,1024]
  const float* A_table = (const float*)d_in[2];   // [64, 8192]
  const float* B_table = (const float*)d_in[3];   // [64, 8192]
  const int*   dom     = (const int*)  d_in[4];   // [16384]
  float*       out     = (float*)d_out;           // [16384,1024]

  // workspace layout: L table (256 KB) | Wh (2 MB) | Wl (2 MB)
  float*          Lt = (float*)d_ws;
  unsigned short* Wh = (unsigned short*)((char*)d_ws + (size_t)N_DOM * OUT_F * 4);
  unsigned short* Wl = Wh + (size_t)OUT_F * IN_F;

  lora_domain_table_kernel<<<N_DOM, 256, 0, stream>>>(A_table, B_table, Lt);
  wsplit_kernel<<<(OUT_F * IN_F / 4) / 256, 256, 0, stream>>>(Wm, Wh, Wl);

  dim3 grid(OUT_F / BN, B_SZ / BM);   // (8, 128): column blocks adjacent so
                                      // x strips + all of W stay L2-resident
  lora_linear_gemm_kernel<<<grid, 256, 0, stream>>>(x, Wh, Wl, dom, Lt, out);
}
